// SimpleRNN_17746804867096
// MI455X (gfx1250) — matrix operand order repaired and checked by reading, not compiled, not run on hardware
//
#include <hip/hip_runtime.h>
#include <math.h>

// Problem dimensions (fixed by the reference)
#define BQ 64      // batch
#define TT 512     // timesteps
#define HH 512     // hidden = embed
#define LL 2       // layers
#define NBLK 32    // persistent workgroups (grid barrier participants)
#define KSTEPS (HH / 32)   // 16 WMMA k-steps per 512-deep GEMM

typedef __attribute__((ext_vector_type(16))) __bf16        v16bf;
typedef __attribute__((ext_vector_type(8)))  float         v8f;
typedef __attribute__((ext_vector_type(4)))  unsigned int  v4u;

union Frag16 {
    v16bf v;
    v4u   q[2];
};

__device__ __forceinline__ unsigned short f2bf(float f) {
    unsigned u = __float_as_uint(f);
    u += 0x7fffu + ((u >> 16) & 1u);     // round-to-nearest-even
    return (unsigned short)(u >> 16);
}
__device__ __forceinline__ float bf2f(unsigned short s) {
    return __uint_as_float(((unsigned)s) << 16);
}

// One 512-deep GEMM tile accumulation with register-resident B fragments.
// Arow is pre-offset per lane with the A lane-half K displacement (kA = 8*hi).
__device__ __forceinline__ v8f gemm512_regW(v8f acc,
                                            const unsigned short* __restrict__ Arow,
                                            const Frag16* w) {
#pragma unroll
    for (int i = 0; i < KSTEPS; ++i) {
        Frag16 a;
        a.q[0] = *(const v4u*)(Arow + i * 32);        // A: K = k0+kA+0..7
        a.q[1] = *(const v4u*)(Arow + i * 32 + 16);   // A: K = k0+kA+16..23
        acc = __builtin_amdgcn_wmma_f32_16x16x32_bf16(
            false, a.v, false, w[i].v, (short)0, acc, false, false);
    }
    return acc;
}

// ---------------------------------------------------------------------------
// Prep kernel 1: convert weights fp32->bf16, zero hidden buffers, reset the
// grid-barrier counter (runs before the persistent kernel each launch, so the
// barrier state is deterministic across graph replays).
// ---------------------------------------------------------------------------
__global__ void prep_weights(const float* __restrict__ Wih_f,
                             const float* __restrict__ Whh_f,
                             unsigned short* __restrict__ Wih_b,
                             unsigned short* __restrict__ Whh_b,
                             unsigned short* __restrict__ h0,
                             unsigned short* __restrict__ h1,
                             unsigned* __restrict__ cnt) {
    const int i = blockIdx.x * blockDim.x + threadIdx.x;   // 0 .. L*H*H-1
    if (i < LL * HH * HH) {
        Wih_b[i] = f2bf(Wih_f[i]);
        Whh_b[i] = f2bf(Whh_f[i]);
    }
    if (i < 2 * BQ * HH) {   // both parity buffers of each state
        h0[i] = 0;
        h1[i] = 0;
    }
    if (i == 0) *cnt = 0u;
}

// ---------------------------------------------------------------------------
// Prep kernel 2: embedding gather + fp32->bf16, laid out [t][b][e] so each
// timestep's A matrix is a contiguous row-major [64 x 512] block.
// ---------------------------------------------------------------------------
__global__ void embed_gather(const int* __restrict__ tokens,
                             const float* __restrict__ emb,
                             unsigned short* __restrict__ Xseq) {
    const int bt = blockIdx.x;          // b*T + t
    const int b  = bt / TT;
    const int t  = bt % TT;
    const int tok = tokens[bt];
    const float* __restrict__ src = emb + (size_t)tok * HH;
    unsigned short* __restrict__ dst = Xseq + ((size_t)t * BQ + b) * HH;
    for (int e = threadIdx.x; e < HH; e += blockDim.x)
        dst[e] = f2bf(src[e]);
}

// ---------------------------------------------------------------------------
// Persistent pipelined RNN kernel: 32 blocks x 256 threads (8 wave32 each).
// Waves 0-3  : layer 0, m-tile = wave,   n-tile = blockIdx.x, timestep s
// Waves 4-7  : layer 1, m-tile = wave-4, n-tile = blockIdx.x, timestep s-1
// One grid barrier per pipeline stage (T+1 stages total).
// Weight fragments for this wave's tile (W_ih + W_hh of its layer) are
// preloaded into 256 VGPRs and stay register-resident for the whole run;
// only the dynamic A operand (x_t / h state) is streamed from L2 per stage.
// ---------------------------------------------------------------------------
__global__ void __launch_bounds__(256, 1) rnn_persistent(
    const unsigned short* __restrict__ Xseq,   // [T][B][H] bf16
    const unsigned short* __restrict__ Wih,    // [L][H][H] bf16
    const unsigned short* __restrict__ Whh,    // [L][H][H] bf16
    const float* __restrict__ bih,             // [L][H] f32
    const float* __restrict__ bhh,             // [L][H] f32
    unsigned short* __restrict__ h0,           // [2][B][H] bf16 (double buffer)
    unsigned short* __restrict__ h1,           // [2][B][H] bf16
    unsigned* __restrict__ cnt) {

    const int lane  = threadIdx.x & 31;
    const int wave  = threadIdx.x >> 5;
    const int grp   = wave >> 2;          // 0 = layer0 group, 1 = layer1 group
    const int mTile = wave & 3;
    const int nTile = blockIdx.x;
    const int hi    = lane >> 4;          // lane half
    const int l16   = lane & 15;
    const int m     = mTile * 16 + l16;   // A row (= batch index) for this lane
    const int n     = nTile * 16 + l16;   // output column for this lane
    const int kA    = hi * 8;             // A lane-half K displacement
    const int kB    = hi * 8;             // B lane-half K displacement

    // ---- preload this wave's layer weights into registers (256 VGPRs) ----
    const unsigned short* __restrict__ W1p =
        Wih + (size_t)grp * HH * HH + (size_t)n * HH + kB;
    const unsigned short* __restrict__ W2p =
        Whh + (size_t)grp * HH * HH + (size_t)n * HH + kB;
    Frag16 w1[KSTEPS], w2[KSTEPS];
#pragma unroll
    for (int i = 0; i < KSTEPS; ++i) {
        w1[i].q[0] = *(const v4u*)(W1p + i * 32);      // B: K = k0+kB+0..7
        w1[i].q[1] = *(const v4u*)(W1p + i * 32 + 16); // B: K = k0+kB+16..23
        w2[i].q[0] = *(const v4u*)(W2p + i * 32);
        w2[i].q[1] = *(const v4u*)(W2p + i * 32 + 16);
    }
    const float bias = bih[grp * HH + n] + bhh[grp * HH + n];

    unsigned barrierGen = 0;
    const unsigned nb = gridDim.x;

    for (int s = 0; s <= TT; ++s) {
        const int pw = s & 1;          // write parity for layer0 at stage s
        const int pr = (s + 1) & 1;    // read parity ((s-1)&1)

        // Wave-uniform activity predicate: layer0 runs s=0..T-1, layer1 s=1..T.
        const bool active = (grp == 0) ? (s < TT) : (s >= 1);
        if (active) {
            const unsigned short* Ax;   // feed-forward input rows
            const unsigned short* Ah;   // recurrent state rows
            unsigned short*       Hout; // destination state buffer
            if (grp == 0) {
                Ax   = Xseq + (size_t)s * BQ * HH;      // x(t=s)
                Ah   = h0 + (size_t)pr * BQ * HH;       // h0(s-1)
                Hout = h0 + (size_t)pw * BQ * HH;       // h0(s)
            } else {
                Ax   = h0 + (size_t)pr * BQ * HH;       // h0(t=s-1), written last stage
                Ah   = h1 + (size_t)pw * BQ * HH;       // h1(s-2)  (parity (s-2)&1 == s&1)
                Hout = h1 + (size_t)pr * BQ * HH;       // h1(s-1)
            }

            // Accumulator initialized with the (per-column) bias.
            v8f acc;
#pragma unroll
            for (int r = 0; r < 8; ++r) acc[r] = bias;

            acc = gemm512_regW(acc, Ax + (size_t)m * HH + kA, w1);
            acc = gemm512_regW(acc, Ah + (size_t)m * HH + kA, w2);

#pragma unroll
            for (int r = 0; r < 8; ++r) {
                // C/D layout: VGPR r -> row (r + 8*hi), col l16
                const int row = mTile * 16 + hi * 8 + r;
                Hout[(size_t)row * HH + n] = f2bf(tanhf(acc[r]));
            }
        }

        // ---- device-wide barrier (monotonic counter, release/acquire) ----
        __threadfence();
        __syncthreads();
        if (threadIdx.x == 0) {
            __hip_atomic_fetch_add(cnt, 1u, __ATOMIC_RELEASE,
                                   __HIP_MEMORY_SCOPE_AGENT);
            ++barrierGen;
            const unsigned target = barrierGen * nb;
            while (__hip_atomic_load(cnt, __ATOMIC_ACQUIRE,
                                     __HIP_MEMORY_SCOPE_AGENT) < target) {
                __builtin_amdgcn_s_sleep(1);
            }
        }
        __syncthreads();
        __threadfence();
    }
}

// ---------------------------------------------------------------------------
// Final FC + sigmoid: out[b] = sigmoid(h1_final[b,:] . W_fc + b_fc).  O == 1.
// ---------------------------------------------------------------------------
__global__ void fc_sigmoid(const unsigned short* __restrict__ hfin,
                           const float* __restrict__ Wfc,
                           const float* __restrict__ bfc,
                           float* __restrict__ out) {
    const int b = threadIdx.x;
    if (b < BQ) {
        float s = 0.f;
        for (int k = 0; k < HH; ++k)
            s += bf2f(hfin[(size_t)b * HH + k]) * Wfc[k];
        s += bfc[0];
        out[b] = 1.f / (1.f + expf(-s));
    }
}

// ---------------------------------------------------------------------------
extern "C" void kernel_launch(void* const* d_in, const int* in_sizes, int n_in,
                              void* d_out, int out_size, void* d_ws, size_t ws_size,
                              hipStream_t stream) {
    const int*   tokens = (const int*)  d_in[0];   // [B,T] int32
    const float* emb    = (const float*)d_in[1];   // [V,E]
    const float* W_ih   = (const float*)d_in[2];   // [L,H,H]
    const float* W_hh   = (const float*)d_in[3];   // [L,H,H]
    const float* b_ih   = (const float*)d_in[4];   // [L,H]
    const float* b_hh   = (const float*)d_in[5];   // [L,H]
    const float* W_fc   = (const float*)d_in[6];   // [O,H] = [1,512]
    const float* b_fc   = (const float*)d_in[7];   // [O]
    float*       out    = (float*)d_out;           // [B,O] = [64]

    // Workspace carve-up (all offsets 256B-aligned)
    char* ws = (char*)d_ws;
    size_t off = 0;
    auto take = [&](size_t bytes) {
        void* p = ws + off;
        off += (bytes + 255) & ~(size_t)255;
        return p;
    };
    unsigned short* Xseq = (unsigned short*)take((size_t)TT * BQ * HH * 2);  // 32 MB
    unsigned short* WihB = (unsigned short*)take((size_t)LL * HH * HH * 2);  // 1 MB
    unsigned short* WhhB = (unsigned short*)take((size_t)LL * HH * HH * 2);  // 1 MB
    unsigned short* h0   = (unsigned short*)take((size_t)2 * BQ * HH * 2);   // 128 KB
    unsigned short* h1   = (unsigned short*)take((size_t)2 * BQ * HH * 2);   // 128 KB
    unsigned*       cnt  = (unsigned*)take(256);
    if (off > ws_size) return;  // workspace too small; nothing safe to do

    // 1) weights -> bf16, zero hidden state, reset barrier counter
    {
        const int nels = LL * HH * HH;              // 524288
        prep_weights<<<(nels + 255) / 256, 256, 0, stream>>>(
            W_ih, W_hh, WihB, WhhB, h0, h1, cnt);
    }
    // 2) embedding gather -> bf16 [t][b][e]
    embed_gather<<<BQ * TT, 256, 0, stream>>>(tokens, emb, Xseq);

    // 3) persistent layer-pipelined WMMA RNN (T+1 stages, 1 barrier each)
    rnn_persistent<<<NBLK, 256, 0, stream>>>(
        Xseq, WihB, WhhB, b_ih, b_hh, h0, h1, cnt);

    // 4) final FC + sigmoid. Layer-1 state for t=T-1 sits at parity (T-1)&1.
    const unsigned short* hfin = h1 + (size_t)((TT - 1) & 1) * BQ * HH;
    fc_sigmoid<<<1, 64, 0, stream>>>(hfin, W_fc, b_fc, out);

    (void)in_sizes; (void)n_in; (void)out_size;
}
